// Head_6794638262366
// MI455X (gfx1250) — compile-verified, matrix operand order checked by reading, not run on hardware
//
#include <hip/hip_runtime.h>

#define N_EMBD 1024
#define HEAD   128
#define SEQ    2048
#define BATCH  8

typedef __attribute__((ext_vector_type(16))) __bf16 v16bf;
typedef __attribute__((ext_vector_type(8)))  float  v8f;

union FragU {
  v16bf bf;
  uint4 q[2];
};

union Pack8 {
  __bf16 h[8];
  uint4  q;
};

union Pack4 {
  __bf16 h[4];
  uint2  q;
};

__device__ __forceinline__ v8f vzero8() {
  v8f z;
  #pragma unroll
  for (int i = 0; i < 8; ++i) z[i] = 0.0f;
  return z;
}

// ---------------------------------------------------------------------------
// Stage 0: W [E,D] f32  ->  Wt [D,E] bf16  (contiguous-K B-fragment loads)
// ---------------------------------------------------------------------------
__global__ void wt_transpose_kernel(const float* __restrict__ W,
                                    __bf16* __restrict__ Wt) {
  int idx = blockIdx.x * blockDim.x + threadIdx.x;
  if (idx >= HEAD * N_EMBD) return;
  int d = idx / N_EMBD;
  int e = idx % N_EMBD;
  Wt[d * N_EMBD + e] = (__bf16)W[e * HEAD + d];
}

// ---------------------------------------------------------------------------
// Stage 1: Q = X*Wq, K = X*Wk, V = X*Wv  (bf16 WMMA, f32 accumulate)
//   Phase 1: block cooperatively converts X tile [16 x 1024] f32 -> LDS bf16
//   Phase 2: 8 waves, wave w computes N-tile w*16; A-frags via ds_load_b128
//   Q,K stored bf16 [B,T,D];  V stored transposed bf16 [B,D,T]
// ---------------------------------------------------------------------------
__global__ void __launch_bounds__(256)
qkv_proj_kernel(const float* __restrict__ X,
                const __bf16* __restrict__ Wqt,
                const __bf16* __restrict__ Wkt,
                const __bf16* __restrict__ Wvt,
                __bf16* __restrict__ Qbf,
                __bf16* __restrict__ Kbf,
                __bf16* __restrict__ Vt) {
  __shared__ __bf16 xtile[16][N_EMBD];          // 32 KB

  const int tid  = threadIdx.x;
  const int tile = blockIdx.x;                  // B * SEQ/16 tiles
  const int b    = tile / (SEQ / 16);
  const int i0   = (tile % (SEQ / 16)) * 16;

  // ---- Phase 1: coalesced f32 load -> bf16 convert -> LDS ----
  {
    const float4* Xblk = (const float4*)(X + ((size_t)b * SEQ + i0) * N_EMBD);
    #pragma unroll
    for (int j = 0; j < 16; ++j) {              // row j, 256 float4 per row
      const float4 x = Xblk[j * 256 + tid];
      Pack4 p;
      p.h[0] = (__bf16)x.x; p.h[1] = (__bf16)x.y;
      p.h[2] = (__bf16)x.z; p.h[3] = (__bf16)x.w;
      *(uint2*)&xtile[j][tid * 4] = p.q;
    }
  }
  __syncthreads();

  // ---- Phase 2: per-wave GEMM ----
  const int lane = tid & 31;
  const int wave = tid >> 5;
  const int n0   = wave * 16;
  const int mrow = lane & 15;
  const int hi   = (lane >> 4) & 1;
  const int koff = hi * 8;

  v8f accQ = vzero8(), accK = vzero8(), accV = vzero8();

  for (int k0 = 0; k0 < N_EMBD; k0 += 32) {
    // A fragment from LDS: two ds_load_b128
    FragU a;
    a.q[0] = *(const uint4*)&xtile[mrow][k0 + koff];
    a.q[1] = *(const uint4*)&xtile[mrow][k0 + koff + 16];

    // B fragments: Wt[D,E] rows are contiguous in K -> two b128 loads each
    const size_t wrow = (size_t)(n0 + mrow) * N_EMBD + k0 + koff;
    FragU bq, bk, bv;
    bq.q[0] = *(const uint4*)(Wqt + wrow);  bq.q[1] = *(const uint4*)(Wqt + wrow + 16);
    bk.q[0] = *(const uint4*)(Wkt + wrow);  bk.q[1] = *(const uint4*)(Wkt + wrow + 16);
    bv.q[0] = *(const uint4*)(Wvt + wrow);  bv.q[1] = *(const uint4*)(Wvt + wrow + 16);

    accQ = __builtin_amdgcn_wmma_f32_16x16x32_bf16(false, a.bf, false, bq.bf, (short)0, accQ, false, false);
    accK = __builtin_amdgcn_wmma_f32_16x16x32_bf16(false, a.bf, false, bk.bf, (short)0, accK, false, false);
    accV = __builtin_amdgcn_wmma_f32_16x16x32_bf16(false, a.bf, false, bv.bf, (short)0, accV, false, false);
  }

  // Q,K: [B,T,D] bf16 (C layout: row m = r + 8*hi, col n = mrow)
  #pragma unroll
  for (int r = 0; r < 8; ++r) {
    const int t = i0 + r + hi * 8;
    const size_t off = ((size_t)b * SEQ + t) * HEAD + n0 + mrow;
    Qbf[off] = (__bf16)accQ[r];
    Kbf[off] = (__bf16)accK[r];
  }
  // V: [B,D,T] bf16 — per lane d fixed, t contiguous over r -> packed b128 store
  Pack8 pv;
  #pragma unroll
  for (int r = 0; r < 8; ++r) pv.h[r] = (__bf16)accV[r];
  const size_t voff = ((size_t)b * HEAD + (n0 + mrow)) * SEQ + i0 + hi * 8;
  *(uint4*)(Vt + voff) = pv.q;
}

// ---------------------------------------------------------------------------
// Stage 2: flash attention, one wave per 16-query tile, key chunks of 32
// ---------------------------------------------------------------------------
__global__ void __launch_bounds__(256)
attn_kernel(const __bf16* __restrict__ Qbf,
            const __bf16* __restrict__ Kbf,
            const __bf16* __restrict__ Vt,
            float* __restrict__ Out) {
  __shared__ __bf16 plds[8][16 * 32];           // P staging: C-layout -> A-layout

  const int lane = threadIdx.x & 31;
  const int wave = threadIdx.x >> 5;
  const int tile = blockIdx.x * 8 + wave;       // B * SEQ/16 tiles total
  const int b    = tile / (SEQ / 16);
  const int i0   = (tile % (SEQ / 16)) * 16;
  const int mrow = lane & 15;
  const int hi   = (lane >> 4) & 1;
  const int koff = hi * 8;
  __bf16* pbuf = &plds[wave][0];

  // Q fragments for all of D=128 (4 x 16x32 A-fragments), held in registers
  FragU qf[4];
  const __bf16* qrow = Qbf + ((size_t)b * SEQ + i0 + mrow) * HEAD;
  #pragma unroll
  for (int c = 0; c < 4; ++c) {
    qf[c].q[0] = *(const uint4*)(qrow + c * 32 + koff);
    qf[c].q[1] = *(const uint4*)(qrow + c * 32 + koff + 16);
  }

  v8f o[8];
  #pragma unroll
  for (int i = 0; i < 8; ++i) o[i] = vzero8();
  float mr[8], lr[8];
  #pragma unroll
  for (int i = 0; i < 8; ++i) { mr[i] = -3.0e38f; lr[i] = 0.0f; }

  const float scale = 0.08838834764831845f;     // 1/sqrt(128)

  const __bf16* Kbase = Kbf + (size_t)b * SEQ * HEAD;
  const __bf16* Vbase = Vt + (size_t)b * HEAD * SEQ;

  for (int j0 = 0; j0 < i0 + 16; j0 += 32) {
    // Prefetch next chunk's K rows / V columns into near caches
    // (global_prefetch_b8; speculative, overlapped with this chunk's WMMAs)
    const int jn = j0 + 32;
    if (jn < i0 + 16) {
      __builtin_prefetch(Kbase + (size_t)(jn + mrow) * HEAD, 0, 3);
      __builtin_prefetch(Kbase + (size_t)(jn + 16 + mrow) * HEAD, 0, 3);
      __builtin_prefetch(Vbase + (size_t)(mrow * 8 + hi * 4) * SEQ + jn, 0, 3);
    }

    // S = Q * K^T for two 16-key tiles (8 bf16 WMMAs)
    v8f s0 = vzero8(), s1 = vzero8();
    #pragma unroll
    for (int c = 0; c < 4; ++c) {
      const __bf16* k0p = Kbase + (size_t)(j0 + mrow) * HEAD + c * 32 + koff;
      const __bf16* k1p = k0p + 16 * HEAD;
      FragU kf0, kf1;
      kf0.q[0] = *(const uint4*)k0p;        kf0.q[1] = *(const uint4*)(k0p + 16);
      kf1.q[0] = *(const uint4*)k1p;        kf1.q[1] = *(const uint4*)(k1p + 16);
      s0 = __builtin_amdgcn_wmma_f32_16x16x32_bf16(false, qf[c].bf, false, kf0.bf, (short)0, s0, false, false);
      s1 = __builtin_amdgcn_wmma_f32_16x16x32_bf16(false, qf[c].bf, false, kf1.bf, (short)0, s1, false, false);
    }

    // scale + causal mask (C layout: row m = r + 8*hi, col n = mrow)
    #pragma unroll
    for (int r = 0; r < 8; ++r) {
      const int qg = i0 + r + hi * 8;
      s0[r] = (j0 + mrow      <= qg) ? s0[r] * scale : -3.0e38f;
      s1[r] = (j0 + 16 + mrow <= qg) ? s1[r] * scale : -3.0e38f;
    }

    // online softmax: per-row max/sum reductions within 16-lane halves
    #pragma unroll
    for (int r = 0; r < 8; ++r) {
      float m = fmaxf(s0[r], s1[r]);
      #pragma unroll
      for (int off = 1; off < 16; off <<= 1) m = fmaxf(m, __shfl_xor(m, off, 32));
      const float mnew = fmaxf(mr[r], m);
      const float corr = __expf(mr[r] - mnew);
      mr[r] = mnew;
      lr[r] *= corr;
      #pragma unroll
      for (int dt = 0; dt < 8; ++dt) o[dt][r] *= corr;

      const float p0 = __expf(s0[r] - mnew);
      const float p1 = __expf(s1[r] - mnew);
      float rs = p0 + p1;
      #pragma unroll
      for (int off = 1; off < 16; off <<= 1) rs += __shfl_xor(rs, off, 32);
      lr[r] += rs;

      const int ml = r + hi * 8;
      pbuf[ml * 32 + mrow]      = (__bf16)p0;
      pbuf[ml * 32 + 16 + mrow] = (__bf16)p1;
    }

    // P as 16x32 A-fragment via LDS (two ds_load_b128)
    FragU pf;
    pf.q[0] = *(const uint4*)(pbuf + mrow * 32 + koff);
    pf.q[1] = *(const uint4*)(pbuf + mrow * 32 + koff + 16);

    // O += P * V   (V transposed [D,T] -> contiguous B-fragment loads)
    #pragma unroll
    for (int dt = 0; dt < 8; ++dt) {
      const __bf16* vp = Vbase + (size_t)(dt * 16 + mrow) * SEQ + j0 + koff;
      FragU vf;
      vf.q[0] = *(const uint4*)vp;          vf.q[1] = *(const uint4*)(vp + 16);
      o[dt] = __builtin_amdgcn_wmma_f32_16x16x32_bf16(false, pf.bf, false, vf.bf, (short)0, o[dt], false, false);
    }
  }

  // epilogue: O /= l, store f32 [B,T,D]
  #pragma unroll
  for (int r = 0; r < 8; ++r) {
    const float inv = 1.0f / lr[r];
    const int t = i0 + r + hi * 8;
    #pragma unroll
    for (int dt = 0; dt < 8; ++dt) {
      Out[((size_t)b * SEQ + t) * HEAD + dt * 16 + mrow] = o[dt][r] * inv;
    }
  }
}

// ---------------------------------------------------------------------------
extern "C" void kernel_launch(void* const* d_in, const int* in_sizes, int n_in,
                              void* d_out, int out_size, void* d_ws, size_t ws_size,
                              hipStream_t stream) {
  const float* X  = (const float*)d_in[0];
  const float* Wq = (const float*)d_in[1];
  const float* Wk = (const float*)d_in[2];
  const float* Wv = (const float*)d_in[3];
  float* Out = (float*)d_out;

  __bf16* ws  = (__bf16*)d_ws;
  __bf16* Wqt = ws;                                   // [D,E] bf16
  __bf16* Wkt = Wqt + (size_t)HEAD * N_EMBD;
  __bf16* Wvt = Wkt + (size_t)HEAD * N_EMBD;
  __bf16* Qbf = Wvt + (size_t)HEAD * N_EMBD;          // [B,T,D] bf16
  __bf16* Kbf = Qbf + (size_t)BATCH * SEQ * HEAD;     // [B,T,D] bf16
  __bf16* Vt  = Kbf + (size_t)BATCH * SEQ * HEAD;     // [B,D,T] bf16

  const int wtElems = HEAD * N_EMBD;
  wt_transpose_kernel<<<(wtElems + 255) / 256, 256, 0, stream>>>(Wq, Wqt);
  wt_transpose_kernel<<<(wtElems + 255) / 256, 256, 0, stream>>>(Wk, Wkt);
  wt_transpose_kernel<<<(wtElems + 255) / 256, 256, 0, stream>>>(Wv, Wvt);

  qkv_proj_kernel<<<BATCH * (SEQ / 16), 256, 0, stream>>>(X, Wqt, Wkt, Wvt, Qbf, Kbf, Vt);

  attn_kernel<<<(BATCH * (SEQ / 16)) / 8, 256, 0, stream>>>(Qbf, Kbf, Vt, Out);
}
